// LSTM_83794811945464
// MI455X (gfx1250) — matrix operand order repaired and checked by reading, not compiled, not run on hardware
//
#include <hip/hip_runtime.h>
#include <hip/hip_bf16.h>

typedef __bf16 bf16;
typedef __attribute__((ext_vector_type(16))) __bf16 v16bf;
typedef __attribute__((ext_vector_type(8)))  __bf16 v8bf;
typedef __attribute__((ext_vector_type(8)))  float  v8f;

#define N_TOT 51200
#define F_IN  64
#define HDIM  256
#define NLAY  3
#define BSZ   256   // number of graphs (ptr has BSZ+1 entries)
#define ODIM  6

// ---------------------------------------------------------------- helpers

// searchsorted(ptr, node, side='right') - 1, clipped to [0, BSZ-1]
__device__ __forceinline__ int graph_idx(const int* __restrict__ ptr, int node) {
    int lo = 0, hi = BSZ + 1;
    while (lo < hi) {
        int mid = (lo + hi) >> 1;
        if (ptr[mid] <= node) lo = mid + 1; else hi = mid;
    }
    int b = lo - 1;
    b = b < 0 ? 0 : b;
    b = b > (BSZ - 1) ? (BSZ - 1) : b;
    return b;
}

// A fragment (16x32 bf16, row-major tile with row stride ldk elements):
// lane l: m = l&15, half = l>>4; elem e at k = kb + 16*half + e.
__device__ __forceinline__ v16bf load_a_frag(const bf16* __restrict__ tile,
                                             int ldk, int kb, int lane) {
    int m = lane & 15, half = lane >> 4;
    const bf16* p = tile + m * ldk + kb + half * 16;
    v8bf lo = *(const v8bf*)(p);
    v8bf hi = *(const v8bf*)(p + 8);
    v16bf a;
#pragma unroll
    for (int e = 0; e < 8; ++e) { a[e] = lo[e]; a[e + 8] = hi[e]; }
    return a;
}

// B fragment (32x16 bf16) from W stored [n][k] row-major (stride ldk):
// lane l: n = l&15, half = l>>4; elem e at k = kb + 16*half + e (contiguous 32B).
__device__ __forceinline__ v16bf load_b_frag(const bf16* __restrict__ wt,
                                             int ldk, int kb, int lane) {
    int nn = lane & 15, half = lane >> 4;
    return *(const v16bf*)(wt + nn * ldk + kb + half * 16);
}

__device__ __forceinline__ v8f wmma_bf16(v16bf a, v16bf b, v8f c) {
    return __builtin_amdgcn_wmma_f32_16x16x32_bf16(false, a, false, b,
                                                   (short)0, c, false, false);
}

__device__ __forceinline__ float sigmoidf_(float x) {
    return 1.0f / (1.0f + __expf(-x));
}

// ---------------------------------------------------------------- weights -> bf16

__global__ void convert_weights(const float* __restrict__ Wenc,
                                const float* __restrict__ Wih,
                                const float* __restrict__ Whh,
                                const float* __restrict__ Wd1,
                                bf16* __restrict__ wenc_t,   // [256][64]
                                bf16* __restrict__ wih_bf,   // [3][1024][256]
                                bf16* __restrict__ whh_bf,   // [3][1024][256]
                                bf16* __restrict__ wd1_t) {  // [64][512]
    int idx = blockIdx.x * 256 + threadIdx.x;
    const int nIH = NLAY * 4 * HDIM * HDIM; // 786432
    if (idx < nIH) {
        wih_bf[idx] = (bf16)Wih[idx];
        whh_bf[idx] = (bf16)Whh[idx];
    }
    if (idx < HDIM * F_IN) {               // transpose [64][256] -> [256][64]
        int h = idx >> 6, k = idx & 63;
        wenc_t[idx] = (bf16)Wenc[k * HDIM + h];
    }
    if (idx < 64 * 2 * HDIM) {             // transpose [512][64] -> [64][512]
        int j = idx >> 9, k = idx & 511;
        wd1_t[idx] = (bf16)Wd1[k * 64 + j];
    }
}

// ---------------------------------------------------------------- encoder
// X = relu(scatter_gm(graph_node) @ W_enc + b_enc), written as bf16 to inp_bf.
// block 128 (4 waves x 16 rows) -> 64 rows per WG, grid 800.

__global__ void encoder_kernel(const float* __restrict__ gnode,
                               const float* __restrict__ gm,
                               const int* __restrict__ ptr,
                               const bf16* __restrict__ wenc_t,
                               const float* __restrict__ benc,
                               bf16* __restrict__ inp_bf) {
    const int LDK = 72;                      // 64 + pad, keeps rows 16B aligned
    __shared__ bf16 smem[4 * 16 * LDK];
    int wave = threadIdx.x >> 5, lane = threadIdx.x & 31;
    int rowbase = blockIdx.x * 64 + wave * 16;
    bf16* At = smem + wave * 16 * LDK;
    int p0 = ptr[0], pend = ptr[BSZ];

    if (lane < 16) {
        int node = rowbase + lane;
        const float* src = gnode + (size_t)node * F_IN;
        bf16* dst = At + lane * LDK;
        bool ok = (node >= p0) && (node < pend);
        int b = graph_idx(ptr, node);
#pragma unroll 8
        for (int c = 0; c < F_IN; ++c) {
            float v = src[c];
            if (ok && c >= F_IN - 10) v = gm[b * 10 + (c - (F_IN - 10))];
            dst[c] = (bf16)v;
        }
    }
    __syncthreads();

    int n = lane & 15, half = lane >> 4;
#pragma unroll
    for (int ht = 0; ht < HDIM / 16; ++ht) {
        v8f acc = {0.f, 0.f, 0.f, 0.f, 0.f, 0.f, 0.f, 0.f};
#pragma unroll
        for (int kc = 0; kc < F_IN / 32; ++kc) {
            v16bf a = load_a_frag(At, LDK, kc * 32, lane);
            v16bf b = load_b_frag(wenc_t + (ht * 16) * F_IN, F_IN, kc * 32, lane);
            acc = wmma_bf16(a, b, acc);
        }
        int col = ht * 16 + n;
        float bias = benc[col];
#pragma unroll
        for (int v = 0; v < 8; ++v) {
            float x = acc[v] + bias;
            x = x > 0.f ? x : 0.f;
            int node = rowbase + v + 8 * half;
            inp_bf[(size_t)node * HDIM + col] = (bf16)x;
        }
    }
}

// ---------------------------------------------------------------- LSTM layer
// gates = inp @ W_ih^T + H @ W_hh^T + b; c,h elementwise; write Hs/Cs fp32 and
// (optionally) gm-scattered next input bf16 in place.
// block 128 = 4 waves, 32 rows per wave (2 M-tiles), grid 400.

__global__ void lstm_layer_kernel(const float* __restrict__ Hin,
                                  const float* __restrict__ Cin,
                                  const bf16* __restrict__ wih,    // [1024][256]
                                  const bf16* __restrict__ whh,    // [1024][256]
                                  const float* __restrict__ bih,   // [1024]
                                  const float* __restrict__ bhh,   // [1024]
                                  const float* __restrict__ gm,
                                  const int* __restrict__ ptr,
                                  bf16* __restrict__ inp_bf,
                                  float* __restrict__ outH,
                                  float* __restrict__ outC,
                                  int write_next) {
    extern __shared__ char dynsmem[];
    const int LDK = HDIM + 8;                 // 264, rows stay 16B aligned
    int wave = threadIdx.x >> 5, lane = threadIdx.x & 31;
    int rowbase = blockIdx.x * 128 + wave * 32;
    bf16* Xs = (bf16*)dynsmem + (size_t)wave * 2 * 32 * LDK;
    bf16* Hs = Xs + 32 * LDK;

    { // stage: each lane owns one of the 32 rows
        int node = rowbase + lane;
        const uint4* src = (const uint4*)(inp_bf + (size_t)node * HDIM);
        uint4* dst = (uint4*)(Xs + lane * LDK);
#pragma unroll 8
        for (int t = 0; t < HDIM * 2 / 16; ++t) dst[t] = src[t];
        const float* hs = Hin + (size_t)node * HDIM;
        bf16* hd = Hs + lane * LDK;
#pragma unroll 8
        for (int c = 0; c < HDIM; ++c) hd[c] = (bf16)hs[c];
    }
    __syncthreads();

    int n = lane & 15, half = lane >> 4;
    int p0 = ptr[0], pend = ptr[BSZ];

    for (int ht = 0; ht < HDIM / 16; ++ht) {
        v8f acc[4][2];
        v8f zero = {0.f, 0.f, 0.f, 0.f, 0.f, 0.f, 0.f, 0.f};
#pragma unroll
        for (int q = 0; q < 4; ++q) { acc[q][0] = zero; acc[q][1] = zero; }

#pragma unroll
        for (int q = 0; q < 4; ++q) {
            const bf16* bihp = wih + (size_t)(q * HDIM + ht * 16) * HDIM;
            const bf16* bhhp = whh + (size_t)(q * HDIM + ht * 16) * HDIM;
#pragma unroll
            for (int kc = 0; kc < HDIM / 32; ++kc) {
                v16bf bi = load_b_frag(bihp, HDIM, kc * 32, lane);
                v16bf bh = load_b_frag(bhhp, HDIM, kc * 32, lane);
#pragma unroll
                for (int mt = 0; mt < 2; ++mt) {
                    v16bf ax = load_a_frag(Xs + mt * 16 * LDK, LDK, kc * 32, lane);
                    v16bf ah = load_a_frag(Hs + mt * 16 * LDK, LDK, kc * 32, lane);
                    acc[q][mt] = wmma_bf16(ax, bi, acc[q][mt]);
                    acc[q][mt] = wmma_bf16(ah, bh, acc[q][mt]);
                }
            }
        }

        int col = ht * 16 + n;
        float bI = bih[0 * HDIM + col] + bhh[0 * HDIM + col];
        float bF = bih[1 * HDIM + col] + bhh[1 * HDIM + col];
        float bG = bih[2 * HDIM + col] + bhh[2 * HDIM + col];
        float bO = bih[3 * HDIM + col] + bhh[3 * HDIM + col];

#pragma unroll
        for (int mt = 0; mt < 2; ++mt) {
#pragma unroll
            for (int v = 0; v < 8; ++v) {
                int node = rowbase + mt * 16 + v + 8 * half;
                size_t off = (size_t)node * HDIM + col;
                float gi = sigmoidf_(acc[0][mt][v] + bI);
                float gf = sigmoidf_(acc[1][mt][v] + bF);
                float gg = tanhf(acc[2][mt][v] + bG);
                float go = sigmoidf_(acc[3][mt][v] + bO);
                float cn = gf * Cin[off] + gi * gg;
                float hn = go * tanhf(cn);
                outC[off] = cn;
                outH[off] = hn;
                if (write_next) {
                    float w = hn;
                    if (col >= HDIM - 10) {
                        bool ok = (node >= p0) && (node < pend);
                        if (ok) {
                            int b = graph_idx(ptr, node);
                            w = gm[b * 10 + (col - (HDIM - 10))];
                        }
                    }
                    inp_bf[off] = (bf16)w;
                }
            }
        }
    }
}

// ---------------------------------------------------------------- decoder
// y = relu(relu([h3|c3] @ W_d1 + b_d1) @ W_d2 + b_d2)
// block 128 = 4 waves x 16 rows, grid 800.

__global__ void decoder_kernel(const float* __restrict__ h3,
                               const float* __restrict__ c3,
                               const bf16* __restrict__ wd1_t, // [64][512]
                               const float* __restrict__ bd1,
                               const float* __restrict__ Wd2,  // [64][6]
                               const float* __restrict__ bd2,
                               float* __restrict__ y) {
    extern __shared__ char dynsmem[];
    const int LDK = 2 * HDIM + 8;            // 520
    int wave = threadIdx.x >> 5, lane = threadIdx.x & 31;
    int rowbase = blockIdx.x * 64 + wave * 16;
    bf16*  St = (bf16*)dynsmem + (size_t)wave * 16 * LDK;
    float* Y1 = (float*)((bf16*)dynsmem + 4 * 16 * LDK) + (size_t)wave * 16 * 64;

    { // stage state rows: lanes 0-15 do the h half, 16-31 the c half
        int r = lane & 15;
        int node = rowbase + r;
        const float* src = (lane < 16 ? h3 : c3) + (size_t)node * HDIM;
        bf16* dst = St + r * LDK + (lane < 16 ? 0 : HDIM);
#pragma unroll 8
        for (int c = 0; c < HDIM; ++c) dst[c] = (bf16)src[c];
    }
    __syncthreads();

    int n = lane & 15, half = lane >> 4;
#pragma unroll
    for (int ct = 0; ct < 4; ++ct) {
        v8f acc = {0.f, 0.f, 0.f, 0.f, 0.f, 0.f, 0.f, 0.f};
#pragma unroll
        for (int kc = 0; kc < (2 * HDIM) / 32; ++kc) {
            v16bf a = load_a_frag(St, LDK, kc * 32, lane);
            v16bf b = load_b_frag(wd1_t + (ct * 16) * (2 * HDIM), 2 * HDIM,
                                  kc * 32, lane);
            acc = wmma_bf16(a, b, acc);
        }
        int j = ct * 16 + n;
        float bias = bd1[j];
#pragma unroll
        for (int v = 0; v < 8; ++v) {
            float x = acc[v] + bias;
            x = x > 0.f ? x : 0.f;
            Y1[(v + 8 * half) * 64 + j] = x;
        }
    }
    __syncthreads();

    if (lane < 16) {
        int node = rowbase + lane;
        const float* yr = Y1 + lane * 64;
#pragma unroll
        for (int o = 0; o < ODIM; ++o) {
            float s = bd2[o];
#pragma unroll 8
            for (int j = 0; j < 64; ++j) s += yr[j] * Wd2[j * ODIM + o];
            s = s > 0.f ? s : 0.f;
            y[(size_t)node * ODIM + o] = s;
        }
    }
}

// ---------------------------------------------------------------- launch

extern "C" void kernel_launch(void* const* d_in, const int* in_sizes, int n_in,
                              void* d_out, int out_size, void* d_ws, size_t ws_size,
                              hipStream_t stream) {
    const float* gm    = (const float*)d_in[0];
    const float* gnode = (const float*)d_in[1];
    const int*   ptr   = (const int*)d_in[2];
    const float* H     = (const float*)d_in[3];
    const float* C     = (const float*)d_in[4];
    const float* Wenc  = (const float*)d_in[5];
    const float* benc  = (const float*)d_in[6];
    const float* Wih   = (const float*)d_in[7];
    const float* bih   = (const float*)d_in[8];
    const float* Whh   = (const float*)d_in[9];
    const float* bhh   = (const float*)d_in[10];
    const float* Wd1   = (const float*)d_in[11];
    const float* bd1   = (const float*)d_in[12];
    const float* Wd2   = (const float*)d_in[13];
    const float* bd2   = (const float*)d_in[14];

    float* out = (float*)d_out;
    char*  ws  = (char*)d_ws;

    // workspace layout (bytes)
    bf16* wih_bf = (bf16*)(ws);                         // 1,572,864
    bf16* whh_bf = (bf16*)(ws + 1572864);               // 1,572,864
    bf16* wenc_t = (bf16*)(ws + 3145728);               //    32,768
    bf16* wd1_t  = (bf16*)(ws + 3178496);               //    65,536
    bf16* inp_bf = (bf16*)(ws + 3244032);               // 26,214,400  (~29.5 MB total)

    convert_weights<<<3072, 256, 0, stream>>>(Wenc, Wih, Whh, Wd1,
                                              wenc_t, wih_bf, whh_bf, wd1_t);

    encoder_kernel<<<800, 128, 0, stream>>>(gnode, gm, ptr, wenc_t, benc, inp_bf);

    const size_t NHD = (size_t)N_TOT * HDIM;
    float* outH = out;
    float* outC = out + (size_t)NLAY * NHD;
    float* yout = out + (size_t)2 * NLAY * NHD;

    const size_t layerSmem = (size_t)4 * 2 * 32 * (HDIM + 8) * sizeof(bf16); // 135,168
    for (int i = 0; i < NLAY; ++i) {
        lstm_layer_kernel<<<400, 128, layerSmem, stream>>>(
            H + (size_t)i * NHD, C + (size_t)i * NHD,
            wih_bf + (size_t)i * 4 * HDIM * HDIM,
            whh_bf + (size_t)i * 4 * HDIM * HDIM,
            bih + (size_t)i * 4 * HDIM, bhh + (size_t)i * 4 * HDIM,
            gm, ptr, inp_bf,
            outH + (size_t)i * NHD, outC + (size_t)i * NHD,
            (i < NLAY - 1) ? 1 : 0);
    }

    const size_t decSmem = (size_t)4 * 16 * (2 * HDIM + 8) * sizeof(bf16)
                         + (size_t)4 * 16 * 64 * sizeof(float);              // 82,944
    decoder_kernel<<<800, 128, decSmem, stream>>>(
        outH + (size_t)(NLAY - 1) * NHD, outC + (size_t)(NLAY - 1) * NHD,
        wd1_t, bd1, Wd2, bd2, yout);
}
